// mamba_block_63230508532236
// MI455X (gfx1250) — compile-verified, matrix operand order checked by reading, not run on hardware
//
#include <hip/hip_runtime.h>

// ---------------------------------------------------------------------------
// Mamba block for gfx1250 (MI455X). bf16 WMMA GEMMs with async-LDS staged
// double-buffered tiles + chunked (3-pass) selective scan.
//   B=2, L=4096, T=B*L=8192, D_MODEL=1024, D_INNER=2048, D_STATE=16, DT_RANK=64
// ---------------------------------------------------------------------------

#define T_TOK   8192
#define DMODEL  1024
#define DINNER  2048
#define DSTATE  16
#define DTRANK  64
#define LSEQ    4096
#define NCHUNK  32
#define CHLEN   128   // LSEQ / NCHUNK

typedef __attribute__((ext_vector_type(16))) __bf16        v16bf;
typedef __attribute__((ext_vector_type(8)))  float         v8f;
typedef __attribute__((ext_vector_type(4)))  unsigned int  v4u;

union BfVec { v16bf v; v4u u[2]; };

#define AS3 __attribute__((address_space(3)))

__device__ __forceinline__ unsigned short f2bf(float f) {
    unsigned u = __float_as_uint(f);
    unsigned r = u + 0x7FFFu + ((u >> 16) & 1u);   // round-to-nearest-even
    return (unsigned short)(r >> 16);
}
__device__ __forceinline__ float b2f(unsigned short h) {
    return __uint_as_float(((unsigned)h) << 16);
}

// 16-byte global -> LDS copy via CDNA5 async path (ASYNCcnt-tracked).
__device__ __forceinline__ void stage16(const unsigned short* g, unsigned short* l) {
    const unsigned lds_off = (unsigned)(unsigned long long)(AS3 void*)l;
    const unsigned long long ga = (unsigned long long)g;
    asm volatile("global_load_async_to_lds_b128 %0, %1, off"
                 :: "v"(lds_off), "v"(ga) : "memory");
}

template<int N>
__device__ __forceinline__ void wait_async() {
#if __has_builtin(__builtin_amdgcn_s_wait_asynccnt)
    __builtin_amdgcn_s_wait_asynccnt((unsigned short)N);
#else
    if constexpr (N == 0) asm volatile("s_wait_asynccnt 0x0" ::: "memory");
    else                  asm volatile("s_wait_asynccnt 0x4" ::: "memory");
#endif
}

// ---------------------------------------------------------------------------
// f32 -> bf16 weight conversion
// ---------------------------------------------------------------------------
__global__ __launch_bounds__(256) void cvt_f2bf(const float* __restrict__ s,
                                                unsigned short* __restrict__ d, int n) {
    int i = blockIdx.x * 256 + threadIdx.x;
    if (i < n) d[i] = f2bf(s[i]);
}

// ---------------------------------------------------------------------------
// RMSNorm: one block per token, output bf16
// ---------------------------------------------------------------------------
__global__ __launch_bounds__(256) void rmsnorm_k(const float* __restrict__ x,
                                                 const float* __restrict__ w,
                                                 unsigned short* __restrict__ h) {
    __shared__ float red[256];
    const int t = blockIdx.x;
    const float* xr = x + (size_t)t * DMODEL;
    float s = 0.0f;
    for (int i = threadIdx.x; i < DMODEL; i += 256) { float v = xr[i]; s += v * v; }
    red[threadIdx.x] = s;
    __syncthreads();
    for (int off = 128; off > 0; off >>= 1) {
        if (threadIdx.x < off) red[threadIdx.x] += red[threadIdx.x + off];
        __syncthreads();
    }
    const float scale = rsqrtf(red[0] * (1.0f / DMODEL) + 1e-5f);
    for (int i = threadIdx.x; i < DMODEL; i += 256)
        h[(size_t)t * DMODEL + i] = f2bf(xr[i] * scale * w[i]);
}

// ---------------------------------------------------------------------------
// Big GEMM: C[M,N] = A[M,K](bf16 rm) * Bw[N,K](bf16 rm)^T
// Block tile 128x128, BK=32, double-buffered async-LDS staging.
// MODE 0: f32 store, 1: bf16 store, 2: bias + softplus, f32 store
// ---------------------------------------------------------------------------
#define SPAD 40

template<int MODE>
__global__ __launch_bounds__(256) void gemm_big_wmma(
    const unsigned short* __restrict__ A,
    const unsigned short* __restrict__ Bw,
    void* __restrict__ Cout,
    const float* __restrict__ bias,
    int M, int N, int K)
{
    __shared__ __align__(16) unsigned short smem[2][2][128 * SPAD];

    const int tiles_n = N / 128;
    const int bm = blockIdx.x / tiles_n;
    const int bn = blockIdx.x - bm * tiles_n;

    const int lane  = threadIdx.x & 31;
    const int rl    = lane & 15;
    const int rsel  = lane >> 4;
    const int klo   = rsel * 8;
    const int waveId = threadIdx.x >> 5;
    const int wm = (waveId & 3) * 32;
    const int wn = (waveId >> 2) * 64;

    const unsigned short* Abase = A + (size_t)(bm * 128) * K;
    const unsigned short* Bbase = Bw + (size_t)(bn * 128) * K;

    auto stage = [&](int buf, int k) {
#pragma unroll
        for (int i = 0; i < 2; ++i) {
            const int c   = threadIdx.x + i * 256;  // 0..511 16B chunks
            const int row = c >> 2;
            const int col = (c & 3) * 8;
            stage16(Abase + (size_t)row * K + k + col, &smem[buf][0][row * SPAD + col]);
            stage16(Bbase + (size_t)row * K + k + col, &smem[buf][1][row * SPAD + col]);
        }
    };

    v8f acc[2][4];
#pragma unroll
    for (int i = 0; i < 2; ++i)
#pragma unroll
        for (int j = 0; j < 4; ++j)
#pragma unroll
            for (int e = 0; e < 8; ++e) acc[i][j][e] = 0.0f;

    const int nk = K / 32;
    stage(0, 0);

    for (int ks = 0; ks < nk; ++ks) {
        const int cur = ks & 1;
        const bool pre = (ks + 1 < nk);
        if (pre) stage(cur ^ 1, (ks + 1) * 32);
        if (pre) wait_async<4>(); else wait_async<0>();
        __syncthreads();

        BfVec a[2];
#pragma unroll
        for (int mt = 0; mt < 2; ++mt) {
            const unsigned short* ap = &smem[cur][0][(wm + mt * 16 + rl) * SPAD + klo];
            a[mt].u[0] = *reinterpret_cast<const v4u*>(ap);
            a[mt].u[1] = *reinterpret_cast<const v4u*>(ap + 16);
        }
#pragma unroll
        for (int t = 0; t < 4; ++t) {
            const unsigned short* bp = &smem[cur][1][(wn + t * 16 + rl) * SPAD + klo];
            BfVec b;
            b.u[0] = *reinterpret_cast<const v4u*>(bp);
            b.u[1] = *reinterpret_cast<const v4u*>(bp + 16);
#pragma unroll
            for (int mt = 0; mt < 2; ++mt)
                acc[mt][t] = __builtin_amdgcn_wmma_f32_16x16x32_bf16(
                    false, a[mt].v, false, b.v, (short)0, acc[mt][t], false, false);
        }
        __syncthreads();
    }

#pragma unroll
    for (int mt = 0; mt < 2; ++mt) {
        const int rowb = bm * 128 + wm + mt * 16 + rsel * 8;
#pragma unroll
        for (int t = 0; t < 4; ++t) {
            const int col = bn * 128 + wn + t * 16 + rl;
#pragma unroll
            for (int r = 0; r < 8; ++r) {
                float v = acc[mt][t][r];
                const size_t idx = (size_t)(rowb + r) * N + col;
                if constexpr (MODE == 2) {
                    v += bias[col];
                    v = (v > 20.0f) ? v : log1pf(__expf(v));
                    ((float*)Cout)[idx] = v;
                } else if constexpr (MODE == 1) {
                    ((unsigned short*)Cout)[idx] = f2bf(v);
                } else {
                    ((float*)Cout)[idx] = v;
                }
            }
        }
    }
}

// ---------------------------------------------------------------------------
// Small GEMM (direct from global) for N=96 x_proj. One wave -> 16 x (16*NT).
// ---------------------------------------------------------------------------
template<int NT>
__global__ __launch_bounds__(256) void gemm_small_wmma(
    const unsigned short* __restrict__ A,
    const unsigned short* __restrict__ Bw,
    float* __restrict__ Cout,
    int M, int N, int K)
{
    const int wave = blockIdx.x * 8 + (threadIdx.x >> 5);
    const int lane = threadIdx.x & 31;
    const int tiles_n = N / (16 * NT);
    const int mt  = wave / tiles_n;
    const int nt0 = (wave - mt * tiles_n) * NT;
    const int rsel = lane >> 4;
    const int rl   = lane & 15;
    const int klo  = rsel * 8;

    const unsigned short* Arow = A + (size_t)(mt * 16 + rl) * K;

    v8f acc[NT];
#pragma unroll
    for (int i = 0; i < NT; ++i)
#pragma unroll
        for (int e = 0; e < 8; ++e) acc[i][e] = 0.0f;

    for (int k = 0; k < K; k += 32) {
        BfVec a;
        const unsigned short* ap = Arow + k + klo;
        a.u[0] = *reinterpret_cast<const v4u*>(ap);
        a.u[1] = *reinterpret_cast<const v4u*>(ap + 16);
#pragma unroll
        for (int t = 0; t < NT; ++t) {
            const unsigned short* bp =
                Bw + (size_t)((nt0 + t) * 16 + rl) * K + k + klo;
            BfVec b;
            b.u[0] = *reinterpret_cast<const v4u*>(bp);
            b.u[1] = *reinterpret_cast<const v4u*>(bp + 16);
            acc[t] = __builtin_amdgcn_wmma_f32_16x16x32_bf16(
                false, a.v, false, b.v, (short)0, acc[t], false, false);
        }
    }

    const int mbase = mt * 16 + rsel * 8;
#pragma unroll
    for (int t = 0; t < NT; ++t) {
        const int col = (nt0 + t) * 16 + rl;
#pragma unroll
        for (int r = 0; r < 8; ++r)
            Cout[(size_t)(mbase + r) * N + col] = acc[t][r];
    }
}

// ---------------------------------------------------------------------------
// Depthwise causal conv (k=4) + bias + SiLU. xi = first half of xz (ld 4096).
// ---------------------------------------------------------------------------
__global__ __launch_bounds__(256) void conv_silu_k(
    const unsigned short* __restrict__ xz,
    const float* __restrict__ cw,
    const float* __restrict__ cb,
    unsigned short* __restrict__ xc)
{
    const int idx = blockIdx.x * 256 + threadIdx.x;   // T * DINNER
    const int d  = idx & (DINNER - 1);
    const int tg = idx >> 11;
    const int t  = tg & (LSEQ - 1);
    const int b  = tg >> 12;
    float acc = cb[d];
    const float* w = cw + d * 4;
#pragma unroll
    for (int j = 0; j < 4; ++j) {
        const int ts = t - 3 + j;
        if (ts >= 0)
            acc += w[j] * b2f(xz[((size_t)(b * LSEQ + ts)) * (2 * DINNER) + d]);
    }
    const float s = acc / (1.0f + __expf(-acc));
    xc[(size_t)tg * DINNER + d] = f2bf(s);
}

// ---------------------------------------------------------------------------
// Slice first DTRANK cols of x_dbl (ld 96) into contiguous bf16 for dt_proj
// ---------------------------------------------------------------------------
__global__ __launch_bounds__(256) void slice_dt_k(const float* __restrict__ xdbl,
                                                  unsigned short* __restrict__ dtin) {
    const int idx = blockIdx.x * 256 + threadIdx.x;   // T * DTRANK
    const int t = idx >> 6;
    const int r = idx & 63;
    dtin[idx] = f2bf(xdbl[(size_t)t * 96 + r]);
}

// ---------------------------------------------------------------------------
// Chunked selective scan (3 passes over NCHUNK chunks of CHLEN steps).
// Thread mapping pass1/3: tid -> d (fast), b, s.  hend/h0 layout:
//   [(b*NCHUNK + s)*DINNER + d]*DSTATE + n
// ---------------------------------------------------------------------------
__global__ __launch_bounds__(256) void scan_part1(
    const float* __restrict__ dt,
    const unsigned short* __restrict__ xc,
    const float* __restrict__ xdbl,
    const float* __restrict__ A_log,
    float* __restrict__ hend,
    float* __restrict__ sumdt)
{
    const int tid = blockIdx.x * 256 + threadIdx.x;   // B*NCHUNK*DINNER
    const int d = tid & (DINNER - 1);
    const int r = tid >> 11;
    const int b = r & 1;
    const int s = r >> 1;

    float Av[DSTATE], h[DSTATE];
#pragma unroll
    for (int n = 0; n < DSTATE; ++n) {
        Av[n] = -__expf(A_log[d * DSTATE + n]);
        h[n] = 0.0f;
    }
    float sacc = 0.0f;
    const int t0 = s * CHLEN;
    for (int tt = 0; tt < CHLEN; ++tt) {
        const size_t tg = (size_t)b * LSEQ + t0 + tt;
        const float dtv = dt[tg * DINNER + d];
        const float xv  = b2f(xc[tg * DINNER + d]);
        const float dtx = dtv * xv;
        const float* bc = xdbl + tg * 96;
        sacc += dtv;
#pragma unroll
        for (int n = 0; n < DSTATE; ++n)
            h[n] = h[n] * __expf(dtv * Av[n]) + dtx * bc[DTRANK + n];
    }
    const size_t idx = (size_t)((b * NCHUNK + s) * DINNER + d);
#pragma unroll
    for (int n = 0; n < DSTATE; ++n) hend[idx * DSTATE + n] = h[n];
    sumdt[idx] = sacc;
}

__global__ __launch_bounds__(256) void scan_part2(
    const float* __restrict__ hend,
    const float* __restrict__ sumdt,
    const float* __restrict__ A_log,
    float* __restrict__ h0)
{
    const int tid = blockIdx.x * 256 + threadIdx.x;   // B*DINNER
    const int d = tid & (DINNER - 1);
    const int b = tid >> 11;

    float Av[DSTATE], carry[DSTATE];
#pragma unroll
    for (int n = 0; n < DSTATE; ++n) {
        Av[n] = -__expf(A_log[d * DSTATE + n]);
        carry[n] = 0.0f;
    }
    for (int s = 0; s < NCHUNK; ++s) {
        const size_t idx = (size_t)((b * NCHUNK + s) * DINNER + d);
#pragma unroll
        for (int n = 0; n < DSTATE; ++n) h0[idx * DSTATE + n] = carry[n];
        const float sd = sumdt[idx];
#pragma unroll
        for (int n = 0; n < DSTATE; ++n)
            carry[n] = carry[n] * __expf(Av[n] * sd) + hend[idx * DSTATE + n];
    }
}

__global__ __launch_bounds__(256) void scan_part3(
    const float* __restrict__ dt,
    const unsigned short* __restrict__ xc,
    const float* __restrict__ xdbl,
    const float* __restrict__ A_log,
    const float* __restrict__ Dp,
    const unsigned short* __restrict__ xz,
    const float* __restrict__ h0,
    unsigned short* __restrict__ y)
{
    const int tid = blockIdx.x * 256 + threadIdx.x;   // B*NCHUNK*DINNER
    const int d = tid & (DINNER - 1);
    const int r = tid >> 11;
    const int b = r & 1;
    const int s = r >> 1;

    float Av[DSTATE], h[DSTATE];
    const size_t cidx = (size_t)((b * NCHUNK + s) * DINNER + d);
#pragma unroll
    for (int n = 0; n < DSTATE; ++n) {
        Av[n] = -__expf(A_log[d * DSTATE + n]);
        h[n] = h0[cidx * DSTATE + n];
    }
    const float Dd = Dp[d];
    const int t0 = s * CHLEN;
    for (int tt = 0; tt < CHLEN; ++tt) {
        const size_t tg = (size_t)b * LSEQ + t0 + tt;
        const float dtv = dt[tg * DINNER + d];
        const float xv  = b2f(xc[tg * DINNER + d]);
        const float dtx = dtv * xv;
        const float* bc = xdbl + tg * 96;
        float yv = 0.0f;
#pragma unroll
        for (int n = 0; n < DSTATE; ++n) {
            h[n] = h[n] * __expf(dtv * Av[n]) + dtx * bc[DTRANK + n];
            yv += h[n] * bc[DTRANK + DSTATE + n];
        }
        yv += xv * Dd;
        const float zv = b2f(xz[tg * (2 * DINNER) + DINNER + d]);
        const float sz = zv / (1.0f + __expf(-zv));
        y[tg * DINNER + d] = f2bf(yv * sz);
    }
}

// ---------------------------------------------------------------------------
// Host-side launcher
// ---------------------------------------------------------------------------
extern "C" void kernel_launch(void* const* d_in, const int* in_sizes, int n_in,
                              void* d_out, int out_size, void* d_ws, size_t ws_size,
                              hipStream_t stream) {
    (void)in_sizes; (void)n_in; (void)out_size; (void)ws_size;

    const float* x         = (const float*)d_in[0];
    const float* norm_w    = (const float*)d_in[1];
    const float* in_proj_w = (const float*)d_in[2];
    const float* conv_w    = (const float*)d_in[3];
    const float* conv_b    = (const float*)d_in[4];
    const float* x_proj_w  = (const float*)d_in[5];
    const float* dt_proj_w = (const float*)d_in[6];
    const float* dt_proj_b = (const float*)d_in[7];
    const float* A_log     = (const float*)d_in[8];
    const float* Dvec      = (const float*)d_in[9];
    const float* out_proj_w= (const float*)d_in[10];
    float* out = (float*)d_out;

    char* p = (char*)d_ws;
    auto take = [&](size_t bytes) -> char* {
        char* r = p;
        p += (bytes + 255) & ~(size_t)255;
        return r;
    };
    unsigned short* wIn  = (unsigned short*)take((size_t)(2*DINNER)*DMODEL * 2);
    unsigned short* wXp  = (unsigned short*)take((size_t)(DTRANK+2*DSTATE)*DINNER * 2);
    unsigned short* wDt  = (unsigned short*)take((size_t)DINNER*DTRANK * 2);
    unsigned short* wOut = (unsigned short*)take((size_t)DMODEL*DINNER * 2);
    unsigned short* hbuf = (unsigned short*)take((size_t)T_TOK*DMODEL * 2);
    unsigned short* xz   = (unsigned short*)take((size_t)T_TOK*2*DINNER * 2);
    unsigned short* xc   = (unsigned short*)take((size_t)T_TOK*DINNER * 2);
    float*          xdbl = (float*)         take((size_t)T_TOK*96 * 4);
    unsigned short* dtin = (unsigned short*)take((size_t)T_TOK*DTRANK * 2);
    float*          dtf  = (float*)         take((size_t)T_TOK*DINNER * 4);
    unsigned short* ybuf = (unsigned short*)take((size_t)T_TOK*DINNER * 2);
    float*          hend = (float*)         take((size_t)2*NCHUNK*DINNER*DSTATE * 4);
    float*          hini = (float*)         take((size_t)2*NCHUNK*DINNER*DSTATE * 4);
    float*          sdt  = (float*)         take((size_t)2*NCHUNK*DINNER * 4);

    {
        int n;
        n = 2*DINNER*DMODEL;            cvt_f2bf<<<(n+255)/256,256,0,stream>>>(in_proj_w,  wIn,  n);
        n = (DTRANK+2*DSTATE)*DINNER;   cvt_f2bf<<<(n+255)/256,256,0,stream>>>(x_proj_w,   wXp,  n);
        n = DINNER*DTRANK;              cvt_f2bf<<<(n+255)/256,256,0,stream>>>(dt_proj_w,  wDt,  n);
        n = DMODEL*DINNER;              cvt_f2bf<<<(n+255)/256,256,0,stream>>>(out_proj_w, wOut, n);
    }

    // 1. RMSNorm
    rmsnorm_k<<<T_TOK, 256, 0, stream>>>(x, norm_w, hbuf);

    // 2. in_proj: [8192,1024] x [4096,1024]^T -> xz bf16
    {
        const int M = T_TOK, N = 2*DINNER, K = DMODEL;
        gemm_big_wmma<1><<<(M/128)*(N/128), 256, 0, stream>>>(hbuf, wIn, xz, nullptr, M, N, K);
    }

    // 3. depthwise causal conv + SiLU -> xc bf16
    conv_silu_k<<<(T_TOK*DINNER)/256, 256, 0, stream>>>(xz, conv_w, conv_b, xc);

    // 4. x_proj: [8192,2048] x [96,2048]^T -> x_dbl f32
    {
        const int M = T_TOK, N = DTRANK + 2*DSTATE, K = DINNER;   // N = 96
        const int waves = (M/16) * (N/32);
        gemm_small_wmma<2><<<waves/8, 256, 0, stream>>>(xc, wXp, xdbl, M, N, K);
    }

    // 5. slice dt columns -> bf16
    slice_dt_k<<<(T_TOK*DTRANK)/256, 256, 0, stream>>>(xdbl, dtin);

    // 6. dt_proj + bias + softplus: [8192,64] x [2048,64]^T -> dt f32
    {
        const int M = T_TOK, N = DINNER, K = DTRANK;
        gemm_big_wmma<2><<<(M/128)*(N/128), 256, 0, stream>>>(dtin, wDt, dtf, dt_proj_b, M, N, K);
    }

    // 7. chunked selective scan (3 passes) -> y bf16
    scan_part1<<<(2*NCHUNK*DINNER)/256, 256, 0, stream>>>(dtf, xc, xdbl, A_log, hend, sdt);
    scan_part2<<<(2*DINNER)/256, 256, 0, stream>>>(hend, sdt, A_log, hini);
    scan_part3<<<(2*NCHUNK*DINNER)/256, 256, 0, stream>>>(dtf, xc, xdbl, A_log, Dvec, xz, hini, ybuf);

    // 8. out_proj: [8192,2048] x [1024,2048]^T -> out f32
    {
        const int M = T_TOK, N = DMODEL, K = DINNER;
        gemm_big_wmma<0><<<(M/128)*(N/128), 256, 0, stream>>>(ybuf, wOut, out, nullptr, M, N, K);
    }
}
